// Attention3d_66692252172535
// MI455X (gfx1250) — matrix operand order repaired and checked by reading, not compiled, not run on hardware
//
#include <hip/hip_runtime.h>
#include <hip/hip_bf16.h>
#include <math.h>

typedef __attribute__((ext_vector_type(16))) __bf16 v16bf;
typedef __attribute__((ext_vector_type(8)))  float  v8f;

#define HEADS 8
#define DH    64
#define WS    16
#define CIN   256
#define INNER 512
#define FDIM  2
#define HDIM  128
#define WDIM  128
#define NPIX  (FDIM*HDIM*WDIM)   // 32768
#define SCALE 0.125f             // DH^-0.5

// ---- CDNA5 async global->LDS copy (ASYNCcnt path), guarded for safety ----
#if defined(__has_builtin)
# if __has_builtin(__builtin_amdgcn_global_load_async_to_lds_b128) && \
     __has_builtin(__builtin_amdgcn_s_wait_asynccnt)
#  define HAVE_ASYNC_LDS 1
# endif
#endif
#ifndef HAVE_ASYNC_LDS
# define HAVE_ASYNC_LDS 0
#endif

#if HAVE_ASYNC_LDS
typedef int v4i __attribute__((vector_size(16)));
typedef __attribute__((address_space(1))) v4i gv4i;   // global-side pointee
typedef __attribute__((address_space(3))) v4i lv4i;   // LDS-side pointee
__device__ __forceinline__ void async_copy16(const unsigned short* g, unsigned short* l) {
    __builtin_amdgcn_global_load_async_to_lds_b128((gv4i*)g, (lv4i*)l, 0, 0);
}
__device__ __forceinline__ void async_wait0() {
    __builtin_amdgcn_s_wait_asynccnt(0);
}
#else
__device__ __forceinline__ void async_copy16(const unsigned short* g, unsigned short* l) {
    *(uint4*)l = *(const uint4*)g;
}
__device__ __forceinline__ void async_wait0() {}
#endif

union Frag {
    v16bf v;
    unsigned short h[16];
    unsigned int   u[8];
};

__device__ __forceinline__ unsigned short f2bf(float x) {
    unsigned int u = __builtin_bit_cast(unsigned int, x);
    unsigned int r = u + 0x7FFFu + ((u >> 16) & 1u);   // round-to-nearest-even
    return (unsigned short)(r >> 16);
}
__device__ __forceinline__ int kpatA(int e, int half) { return (e < 8 ? e : e + 8) + 8 * half; }
__device__ __forceinline__ int kpatB(int e, int half) { return (e < 8 ? e : e + 8) + 8 * half; }

// ---------------------------------------------------------------------------
// Kernel 1: fused Q/K/V projection.  Y[p][n] = sum_c X[c][p] * Wcat[n][c]
// Wcat rows: [0,512)=wq -> q, [512,1024)=wkv[0:512] -> k, [1024,1536) -> v
// Output: bf16, layout [pixel][512] per buffer.
// ---------------------------------------------------------------------------
__global__ __launch_bounds__(256)
void qkv_proj_kernel(const float* __restrict__ x,
                     const float* __restrict__ wq,
                     const float* __restrict__ wkv,
                     unsigned short* __restrict__ qbuf,
                     unsigned short* __restrict__ kbuf,
                     unsigned short* __restrict__ vbuf)
{
    __shared__ unsigned short ldsA[128][32];   // [m][k] bf16, 8 KB
    __shared__ unsigned short ldsB[64][32];    // [n][k] bf16, 4 KB

    const int tid  = threadIdx.x;
    const int wave = tid >> 5, lane = tid & 31;
    const int col  = lane & 15, half = lane >> 4;
    const int m0 = blockIdx.x * 128;
    const int n0 = blockIdx.y * 64;

    const v8f vzero = {0.f,0.f,0.f,0.f,0.f,0.f,0.f,0.f};
    v8f acc[4];
    for (int t = 0; t < 4; ++t) acc[t] = vzero;

    for (int k0 = 0; k0 < CIN; k0 += 32) {
        // Stage A tile (32 K x 128 M), coalesced over m, transpose to [m][k]
        for (int i = 0; i < 16; ++i) {
            int idx = tid + i * 256;
            int mm = idx & 127, kk = idx >> 7;
            ldsA[mm][kk] = f2bf(x[(size_t)(k0 + kk) * NPIX + (m0 + mm)]);
        }
        // Prefetch next X tile (streams once from HBM)
        if (k0 + 32 < CIN)
            __builtin_prefetch(&x[(size_t)(k0 + 32 + (tid >> 7)) * NPIX + (m0 + (tid & 127))], 0, 1);
        // Stage B tile [n][k], coalesced over k
        for (int i = 0; i < 8; ++i) {
            int idx = tid * 8 + i;
            int n = idx >> 5, kk = idx & 31;
            int ng = n0 + n;
            float wv = (ng < 512) ? wq[(size_t)ng * CIN + (k0 + kk)]
                                  : wkv[(size_t)(ng - 512) * CIN + (k0 + kk)];
            ldsB[n][kk] = f2bf(wv);
        }
        __syncthreads();

        Frag a;
        for (int e2 = 0; e2 < 8; ++e2)
            a.u[e2] = *(const unsigned int*)&ldsA[wave * 16 + col][kpatA(2 * e2, half)];
        for (int t = 0; t < 4; ++t) {
            Frag b;
            for (int e2 = 0; e2 < 8; ++e2)
                b.u[e2] = *(const unsigned int*)&ldsB[t * 16 + col][kpatB(2 * e2, half)];
            acc[t] = __builtin_amdgcn_wmma_f32_16x16x32_bf16(
                false, a.v, false, b.v, (short)0, acc[t], false, false);
        }
        __syncthreads();
    }

    const int group = n0 >> 9;  // 0=q, 1=k, 2=v (block never straddles groups)
    unsigned short* buf = (group == 0) ? qbuf : (group == 1 ? kbuf : vbuf);
    const int nbase = n0 & 511;
    for (int t = 0; t < 4; ++t) {
        int n = nbase + t * 16 + col;
        for (int r = 0; r < 8; ++r) {
            int p = m0 + wave * 16 + r + 8 * half;
            buf[(size_t)p * INNER + n] = f2bf(acc[t][r]);
        }
    }
}

// ---------------------------------------------------------------------------
// Kernel 2: windowed flash attention.
// grid.x = head*window (512), grid.y = q strip of 128 tokens (4).
// Each wave owns 16 query rows; loop over kv tiles of 64 tokens.
// K tile staged via async global->LDS DMA; V staged manually (transposed).
// ---------------------------------------------------------------------------
__global__ __launch_bounds__(256)
void attn_kernel(const unsigned short* __restrict__ qbuf,
                 const unsigned short* __restrict__ kbuf,
                 const unsigned short* __restrict__ vbuf,
                 unsigned short* __restrict__ obuf)
{
    __shared__ unsigned short ldsK[64][64];     // [token][d]   8 KB
    __shared__ unsigned short ldsV[64][64];     // [d][token]   8 KB (transposed)
    __shared__ unsigned short ldsP[8][64][16];  // [wave][kvtok][qrow] 16 KB

    const int tid  = threadIdx.x;
    const int wave = tid >> 5, lane = tid & 31;
    const int col  = lane & 15, half = lane >> 4;

    const int hw   = blockIdx.x;
    const int head = hw >> 6;
    const int win  = hw & 63;
    const int xw   = win >> 3, yw = win & 7;
    const int t0   = blockIdx.y * 128 + wave * 16;
    const int cb   = head * DH;

    auto pixel_of = [&](int t) -> int {
        int f = t >> 8, rem = t & 255;
        int w1 = rem >> 4, w2 = rem & 15;
        return f * (HDIM * WDIM) + (xw * WS + w1) * WDIM + (yw * WS + w2);
    };

    // Q fragments: A layout, K-dim = dh (2 steps of 32); resident all loop.
    Frag aq[2];
    {
        size_t base = (size_t)pixel_of(t0 + col) * INNER + cb;
        for (int ks = 0; ks < 2; ++ks)
            for (int e2 = 0; e2 < 8; ++e2)
                aq[ks].u[e2] = *(const unsigned int*)(qbuf + base + ks * 32 + kpatA(2 * e2, half));
    }

    const v8f vzero = {0.f,0.f,0.f,0.f,0.f,0.f,0.f,0.f};
    float mrow[8], lrow[8];
    v8f accO[4];
    for (int r = 0; r < 8; ++r) { mrow[r] = -1e30f; lrow[r] = 0.f; }
    for (int t = 0; t < 4; ++t) accO[t] = vzero;

    for (int j = 0; j < 8; ++j) {
        // ---- stage K via async DMA, V manually transposed ----
        {
            int row = tid >> 2;       // kv token within tile
            int part = tid & 3;       // d chunk of 16
            size_t gb = (size_t)pixel_of(j * 64 + row) * INNER + cb + part * 16;
            async_copy16(kbuf + gb,     &ldsK[row][part * 16]);
            async_copy16(kbuf + gb + 8, &ldsK[row][part * 16 + 8]);
            const uint4* gv = (const uint4*)(vbuf + gb);
            uint4 va = gv[0], vb = gv[1];
            unsigned int vv[8] = {va.x, va.y, va.z, va.w, vb.x, vb.y, vb.z, vb.w};
            for (int q = 0; q < 8; ++q) {
                int d = part * 16 + q * 2;
                ldsV[d][row]     = (unsigned short)(vv[q] & 0xFFFFu);
                ldsV[d + 1][row] = (unsigned short)(vv[q] >> 16);
            }
        }
        async_wait0();         // each wave's async copies complete ...
        __syncthreads();       // ... before any wave reads the shared tile

        // ---- S = scale * Q K^T  (16 x 64 per wave) ----
        v8f s[4];
        for (int t = 0; t < 4; ++t) s[t] = vzero;
        for (int t = 0; t < 4; ++t)
            for (int ks = 0; ks < 2; ++ks) {
                Frag b;
                for (int e2 = 0; e2 < 8; ++e2)
                    b.u[e2] = *(const unsigned int*)&ldsK[t * 16 + col][ks * 32 + kpatB(2 * e2, half)];
                s[t] = __builtin_amdgcn_wmma_f32_16x16x32_bf16(
                    false, aq[ks].v, false, b.v, (short)0, s[t], false, false);
            }

        // ---- online softmax (rows live in lane-halves of the C layout) ----
        for (int r = 0; r < 8; ++r) {
            float m = -1e30f;
            for (int t = 0; t < 4; ++t) { s[t][r] *= SCALE; m = fmaxf(m, s[t][r]); }
            for (int off = 1; off < 16; off <<= 1) m = fmaxf(m, __shfl_xor(m, off, 32));
            float mn = fmaxf(mrow[r], m);
            float corr = __expf(mrow[r] - mn);
            mrow[r] = mn;
            float psum = 0.f;
            for (int t = 0; t < 4; ++t) {
                float p = __expf(s[t][r] - mn);
                s[t][r] = p;
                psum += p;
            }
            for (int off = 1; off < 16; off <<= 1) psum += __shfl_xor(psum, off, 32);
            lrow[r] = lrow[r] * corr + psum;
            for (int t = 0; t < 4; ++t) accO[t][r] *= corr;
        }

        // ---- re-layout P: C-fragment -> LDS -> A-fragment (wave-private) ----
        for (int t = 0; t < 4; ++t)
            for (int r = 0; r < 8; ++r)
                ldsP[wave][t * 16 + col][r + 8 * half] = f2bf(s[t][r]);

        // ---- O += P V ----
        for (int ks = 0; ks < 2; ++ks) {
            Frag ap;
            for (int e = 0; e < 16; ++e)
                ap.h[e] = ldsP[wave][ks * 32 + kpatA(e, half)][col];
            for (int t = 0; t < 4; ++t) {
                Frag bv;
                for (int e2 = 0; e2 < 8; ++e2)
                    bv.u[e2] = *(const unsigned int*)&ldsV[t * 16 + col][ks * 32 + kpatB(2 * e2, half)];
                accO[t] = __builtin_amdgcn_wmma_f32_16x16x32_bf16(
                    false, ap.v, false, bv.v, (short)0, accO[t], false, false);
            }
        }
        __syncthreads();   // protect ldsK/ldsV before next stage
    }

    // ---- normalize + store O as bf16 [pixel][512] ----
    for (int r = 0; r < 8; ++r) {
        float inv = 1.0f / lrow[r];
        size_t base = (size_t)pixel_of(t0 + r + 8 * half) * INNER + cb;
        for (int t = 0; t < 4; ++t)
            obuf[base + t * 16 + col] = f2bf(accO[t][r] * inv);
    }
}

// ---------------------------------------------------------------------------
// Kernel 3: output projection.  out[n][p] = sum_i O[p][i]*wo[n][i] + bo[n]
// A tile (bf16, conversion-free) staged via async global->LDS DMA.
// ---------------------------------------------------------------------------
__global__ __launch_bounds__(256)
void out_proj_kernel(const unsigned short* __restrict__ obuf,
                     const float* __restrict__ wo,
                     const float* __restrict__ bo,
                     float* __restrict__ out)
{
    __shared__ unsigned short ldsA[128][32];  // [m][k] bf16, 8 KB
    __shared__ unsigned short ldsB[64][32];   // [n][k] bf16, 4 KB

    const int tid  = threadIdx.x;
    const int wave = tid >> 5, lane = tid & 31;
    const int col  = lane & 15, half = lane >> 4;
    const int m0 = blockIdx.x * 128;
    const int n0 = blockIdx.y * 64;

    const v8f vzero = {0.f,0.f,0.f,0.f,0.f,0.f,0.f,0.f};
    v8f acc[4];
    for (int t = 0; t < 4; ++t) acc[t] = vzero;

    for (int k0 = 0; k0 < INNER; k0 += 32) {
        // Stage A tile via async DMA: 128 rows x 32 k = 512 x 16B chunks
        for (int i = 0; i < 2; ++i) {
            int idx = tid * 2 + i;
            int row = idx >> 2, part = idx & 3;
            async_copy16(obuf + (size_t)(m0 + row) * INNER + k0 + part * 8,
                         &ldsA[row][part * 8]);
        }
        // Stage B tile [n][k] (fp32 -> bf16 conversion, manual)
        for (int i = 0; i < 8; ++i) {
            int idx = tid * 8 + i;
            int n = idx >> 5, kk = idx & 31;
            ldsB[n][kk] = f2bf(wo[(size_t)(n0 + n) * INNER + k0 + kk]);
        }
        async_wait0();
        __syncthreads();

        Frag a;
        for (int e2 = 0; e2 < 8; ++e2)
            a.u[e2] = *(const unsigned int*)&ldsA[wave * 16 + col][kpatA(2 * e2, half)];
        for (int t = 0; t < 4; ++t) {
            Frag b;
            for (int e2 = 0; e2 < 8; ++e2)
                b.u[e2] = *(const unsigned int*)&ldsB[t * 16 + col][kpatB(2 * e2, half)];
            acc[t] = __builtin_amdgcn_wmma_f32_16x16x32_bf16(
                false, a.v, false, b.v, (short)0, acc[t], false, false);
        }
        __syncthreads();
    }

    for (int t = 0; t < 4; ++t) {
        int n = n0 + t * 16 + col;
        float bias = bo[n];
        for (int r = 0; r < 8; ++r) {
            int p = m0 + wave * 16 + r + 8 * half;
            out[(size_t)n * NPIX + p] = acc[t][r] + bias;
        }
    }
}

// ---------------------------------------------------------------------------
extern "C" void kernel_launch(void* const* d_in, const int* in_sizes, int n_in,
                              void* d_out, int out_size, void* d_ws, size_t ws_size,
                              hipStream_t stream)
{
    const float* x   = (const float*)d_in[0];
    const float* wq  = (const float*)d_in[1];
    const float* wkv = (const float*)d_in[2];
    const float* wo  = (const float*)d_in[3];
    const float* bo  = (const float*)d_in[4];
    float* out = (float*)d_out;

    const size_t bufElems = (size_t)NPIX * INNER;          // 16.7M bf16 each
    unsigned short* qbuf = (unsigned short*)d_ws;
    unsigned short* kbuf = qbuf + bufElems;
    unsigned short* vbuf = kbuf + bufElems;
    unsigned short* obuf = vbuf + bufElems;                // total 128 MB (fits L2)

    qkv_proj_kernel<<<dim3(NPIX / 128, 1536 / 64), 256, 0, stream>>>(
        x, wq, wkv, qbuf, kbuf, vbuf);
    attn_kernel<<<dim3(HEADS * 64, 4), 256, 0, stream>>>(
        qbuf, kbuf, vbuf, obuf);
    out_proj_kernel<<<dim3(NPIX / 128, 256 / 64), 256, 0, stream>>>(
        obuf, wo, bo, out);
}
